// SlotAttentionVideo_78305843741000
// MI455X (gfx1250) — matrix operand order repaired and checked by reading, not compiled, not run on hardware
//
#include <hip/hip_runtime.h>
#include <hip/hip_bf16.h>
#include <math.h>

typedef __attribute__((ext_vector_type(16))) _Float16 v16h;
typedef __attribute__((ext_vector_type(8)))  _Float16 v8h;
typedef __attribute__((ext_vector_type(4)))  _Float16 v4h;
typedef __attribute__((ext_vector_type(8)))  float    v8f;

#define BB   64
#define NN   4096
#define DINC 128
#define EEC  64
#define EPSA 1e-8f

// ---------------------------------------------------------------------------
// K1: fused LayerNorm(inputs) -> kk = xW_k * 128^-0.5 (f16), vv = xW_v (f16)
// 4 waves / block, 16 rows per wave, WMMA f32_16x16x32_f16.
// ldsW holds W pre-swizzled in WMMA B-fragment order (contiguous 16-half
// fragments -> ds_load_b128). After A-fragments are built, each wave reuses
// its ldsA region as a C-staging tile so global stores are coalesced b128.
// ---------------------------------------------------------------------------
__global__ __launch_bounds__(128)
void k_ln_gemm(const float* __restrict__ x,
               const float* __restrict__ lng, const float* __restrict__ lnb,
               const float* __restrict__ Wk, const float* __restrict__ Wv,
               _Float16* __restrict__ kkh, _Float16* __restrict__ vvh)
{
    __shared__ _Float16 ldsW[128 * 128];  // 32 KB, B-fragment-major swizzle
    __shared__ _Float16 ldsA[64 * 128];   // 16 KB: LN rows, then C staging

    const int tid  = threadIdx.x;
    const int w    = tid >> 5;      // wave 0..3
    const int lane = tid & 31;
    const long row0 = (long)blockIdx.x * 64;

    // ---- LayerNorm 16 rows per wave into ldsA (f16, row-major) ----
    for (int r = 0; r < 16; ++r) {
        const int wr = w * 16 + r;
        const float* xr = x + (row0 + wr) * DINC;
        float v0[4];
        float s = 0.f, s2 = 0.f;
        #pragma unroll
        for (int j = 0; j < 4; ++j) {
            float t = xr[lane * 4 + j];
            v0[j] = t; s += t; s2 += t * t;
        }
        #pragma unroll
        for (int off = 16; off > 0; off >>= 1) {
            s  += __shfl_xor(s,  off, 32);
            s2 += __shfl_xor(s2, off, 32);
        }
        const float mean = s * (1.f / 128.f);
        const float var  = s2 * (1.f / 128.f) - mean * mean;
        const float inv  = rsqrtf(var + 1e-5f);
        #pragma unroll
        for (int j = 0; j < 4; ++j) {
            const int col = lane * 4 + j;
            ldsA[wr * 128 + col] =
                (_Float16)((v0[j] - mean) * inv * lng[col] + lnb[col]);
        }
    }
    // stage Wk swizzled into B-fragment order:
    //   ldsW[((t*4+c)*32 + l)*16 + h] = W[(c*32 + (h<8 ? (l>=16?8:0) + h : 16 + (l>=16?8:0) + h-8)) * 128 + t*16 + (l&15)]
    for (int idx = tid; idx < 128 * 128; idx += 128) {
        const int h = idx & 15;
        const int l = (idx >> 4) & 31;
        const int c = (idx >> 9) & 3;
        const int t = idx >> 11;
        const int kk_ = c * 32 + ((l >> 4) << 3) + (h & 7) + ((h >> 3) << 4);
        const int nn_ = t * 16 + (l & 15);
        ldsW[idx] = (_Float16)Wk[kk_ * 128 + nn_];
    }
    __syncthreads();

    // ---- A fragments: 16-bit A 16x32 lane layout, vector LDS loads ----
    v16h afrag[4];
    {
        const int m  = lane & 15;
        const int kb = (lane >> 4) * 8;
        const _Float16* abase = &ldsA[(w * 16 + m) * 128];
        #pragma unroll
        for (int c = 0; c < 4; ++c) {
            const v8h lo = *(const v8h*)(abase + c * 32 + kb);
            const v8h hi = *(const v8h*)(abase + c * 32 + kb + 16);
            #pragma unroll
            for (int j = 0; j < 8; ++j) {
                afrag[c][j]     = lo[j];
                afrag[c][8 + j] = hi[j];
            }
        }
    }
    const int ncol = lane & 15;
    const int mhi  = (lane >> 4) * 8;
    const float kscale = 0.0883883476483184f; // 128^-0.5
    _Float16* ldsC = &ldsA[w * 16 * 128];     // wave-private, A rows now dead

    // ---- phase 1: kk ----
    #pragma unroll 2
    for (int t = 0; t < 8; ++t) {
        v8f acc = {};
        #pragma unroll
        for (int c = 0; c < 4; ++c) {
            const v16h bfrag = *(const v16h*)&ldsW[((t * 4 + c) * 32 + lane) * 16];
            acc = __builtin_amdgcn_wmma_f32_16x16x32_f16(
                false, afrag[c], false, bfrag, (short)0, acc, false, false);
        }
        #pragma unroll
        for (int r = 0; r < 8; ++r)
            ldsC[(r + mhi) * 128 + t * 16 + ncol] = (_Float16)(acc[r] * kscale);
    }
    {   // coalesced write-back: each lane copies half a row (128 B)
        const int r16 = lane >> 1;
        const int cb  = (lane & 1) * 64;
        const uint4* src = (const uint4*)&ldsC[r16 * 128 + cb];
        uint4* dst = (uint4*)(kkh + (row0 + w * 16 + r16) * 128 + cb);
        #pragma unroll
        for (int q = 0; q < 8; ++q) dst[q] = src[q];
    }
    __syncthreads();
    for (int idx = tid; idx < 128 * 128; idx += 128) {
        const int h = idx & 15;
        const int l = (idx >> 4) & 31;
        const int c = (idx >> 9) & 3;
        const int t = idx >> 11;
        const int kk_ = c * 32 + ((l >> 4) << 3) + (h & 7) + ((h >> 3) << 4);
        const int nn_ = t * 16 + (l & 15);
        ldsW[idx] = (_Float16)Wv[kk_ * 128 + nn_];
    }
    __syncthreads();

    // ---- phase 2: vv ----
    #pragma unroll 2
    for (int t = 0; t < 8; ++t) {
        v8f acc = {};
        #pragma unroll
        for (int c = 0; c < 4; ++c) {
            const v16h bfrag = *(const v16h*)&ldsW[((t * 4 + c) * 32 + lane) * 16];
            acc = __builtin_amdgcn_wmma_f32_16x16x32_f16(
                false, afrag[c], false, bfrag, (short)0, acc, false, false);
        }
        #pragma unroll
        for (int r = 0; r < 8; ++r)
            ldsC[(r + mhi) * 128 + t * 16 + ncol] = (_Float16)acc[r];
    }
    {
        const int r16 = lane >> 1;
        const int cb  = (lane & 1) * 64;
        const uint4* src = (const uint4*)&ldsC[r16 * 128 + cb];
        uint4* dst = (uint4*)(vvh + (row0 + w * 16 + r16) * 128 + cb);
        #pragma unroll
        for (int q = 0; q < 8; ++q) dst[q] = src[q];
    }
}

// ---------------------------------------------------------------------------
// K2: p = LN(protos) @ Wp   (64x64)
// ---------------------------------------------------------------------------
__global__ void k_protos(const float* __restrict__ protos,
                         const float* __restrict__ g, const float* __restrict__ b,
                         const float* __restrict__ Wp, float* __restrict__ p_out)
{
    const int c = threadIdx.x; // 64 threads
    if (c >= 64) return;
    float row[64];
    float s = 0.f, s2 = 0.f;
    for (int k = 0; k < 64; ++k) {
        float t = protos[c * 64 + k];
        row[k] = t; s += t; s2 += t * t;
    }
    const float m = s / 64.f;
    const float var = s2 / 64.f - m * m;
    const float inv = rsqrtf(var + 1e-5f);
    for (int k = 0; k < 64; ++k) row[k] = (row[k] - m) * inv * g[k] + b[k];
    for (int i = 0; i < 64; ++i) {
        float a = 0.f;
        for (int k = 0; k < 64; ++k) a += row[k] * Wp[k * 64 + i];
        p_out[c * 64 + i] = a;
    }
}

// ---------------------------------------------------------------------------
// K_init: reparameterized slots_ext / logits_pi
// ---------------------------------------------------------------------------
__global__ void k_init(const float* __restrict__ ext_mu, const float* __restrict__ ext_ls,
                       const float* __restrict__ ext_noise,
                       const float* __restrict__ int_mu, const float* __restrict__ int_ls,
                       const float* __restrict__ int_noise,
                       float* __restrict__ slots_ext, float* __restrict__ logits_pi)
{
    const int t = blockIdx.x * blockDim.x + threadIdx.x; // 512*64
    if (t >= 512 * 64) return;
    const int e = t & 63;
    slots_ext[t] = ext_mu[e] + expf(ext_ls[e]) * ext_noise[t];
    logits_pi[t] = int_mu[e] + expf(int_ls[e]) * int_noise[t];
}

// ---------------------------------------------------------------------------
// K3a: gumbel softmax -> slots_int -> slots -> q ; zero colsum
// one wave per (b,s); grid = B blocks x 256 threads
// ---------------------------------------------------------------------------
__global__ __launch_bounds__(256)
void k_slots_q(const float* __restrict__ logits_pi,
               const float* __restrict__ gumbel_i,
               const int* __restrict__ tau_p,
               const float* __restrict__ p,
               const float* __restrict__ slots_ext,
               const float* __restrict__ slots_bck,
               const float* __restrict__ Wq,
               float* __restrict__ slots_int_ws,
               float* __restrict__ q_ws,
               float* __restrict__ colsum)
{
    __shared__ float pi[8 * 64];
    __shared__ float slot[8 * 128];
    const int tid = threadIdx.x;
    const int s = tid >> 5, lane = tid & 31;
    const int b = blockIdx.x;
    const int t = b * 8 + s;

    if (lane == 0) colsum[t] = 0.f;
    const float inv_tau = 1.f / (float)tau_p[0];

    float l0 = (logits_pi[t * 64 + lane]      + gumbel_i[t * 64 + lane])      * inv_tau;
    float l1 = (logits_pi[t * 64 + lane + 32] + gumbel_i[t * 64 + lane + 32]) * inv_tau;
    float mx = fmaxf(l0, l1);
    #pragma unroll
    for (int off = 16; off > 0; off >>= 1) mx = fmaxf(mx, __shfl_xor(mx, off, 32));
    const float e0 = expf(l0 - mx), e1 = expf(l1 - mx);
    float sum = e0 + e1;
    #pragma unroll
    for (int off = 16; off > 0; off >>= 1) sum += __shfl_xor(sum, off, 32);
    const float inv = 1.f / sum;
    pi[s * 64 + lane]      = e0 * inv;
    pi[s * 64 + lane + 32] = e1 * inv;
    __syncthreads();

    #pragma unroll
    for (int half = 0; half < 2; ++half) {
        const int i = lane + half * 32;
        float a = 0.f;
        for (int c = 0; c < 64; ++c) a += pi[s * 64 + c] * p[c * 64 + i];
        slots_int_ws[t * 64 + i] = a;                 // pre-concat slots_int
        slot[s * 128 + i]      = (s == 0) ? slots_bck[b * 128 + i]      : a;
        slot[s * 128 + 64 + i] = (s == 0) ? slots_bck[b * 128 + 64 + i] : slots_ext[t * 64 + i];
    }
    __syncthreads();

    #pragma unroll
    for (int k4 = 0; k4 < 4; ++k4) {
        const int d2 = lane + k4 * 32;
        float a = 0.f;
        for (int d = 0; d < 128; ++d) a += slot[s * 128 + d] * Wq[d * 128 + d2];
        q_ws[t * 128 + d2] = a;
    }
}

// ---------------------------------------------------------------------------
// K3b: attn_logits = kk . q, softmax over S, write attn_vis, column sums
// grid (16, B), 256 threads, one n per thread, 128-bit f16 loads
// ---------------------------------------------------------------------------
__global__ __launch_bounds__(256)
void k_attn(const _Float16* __restrict__ kkh, const float* __restrict__ q_ws,
            float* __restrict__ attn_out, float* __restrict__ colsum)
{
    __shared__ float ldsq[8 * 128];
    __shared__ float scs[8];
    const int tid = threadIdx.x;
    const int b = blockIdx.y;
    const long n = (long)blockIdx.x * 256 + tid;

    if (tid < 8) scs[tid] = 0.f;
    for (int idx = tid; idx < 1024; idx += 256)
        ldsq[idx] = q_ws[b * 1024 + idx];
    __syncthreads();

    const v8h* kr = (const v8h*)(kkh + ((long)b * NN + n) * 128);
    float acc[8] = {};
    for (int d8 = 0; d8 < 16; ++d8) {
        const v8h kv = kr[d8];
        #pragma unroll
        for (int j = 0; j < 8; ++j) {
            const float kvf = (float)kv[j];
            const int d = d8 * 8 + j;
            #pragma unroll
            for (int s = 0; s < 8; ++s) acc[s] += kvf * ldsq[s * 128 + d];
        }
    }
    float mx = acc[0];
    #pragma unroll
    for (int s = 1; s < 8; ++s) mx = fmaxf(mx, acc[s]);
    float sum = 0.f;
    #pragma unroll
    for (int s = 0; s < 8; ++s) { acc[s] = expf(acc[s] - mx); sum += acc[s]; }
    const float inv = 1.f / sum;
    float* ao = attn_out + ((long)b * NN + n) * 8;
    #pragma unroll
    for (int s = 0; s < 8; ++s) {
        const float v = acc[s] * inv;
        ao[s] = v;
        atomicAdd(&scs[s], v + EPSA);
    }
    __syncthreads();
    if (tid < 8) atomicAdd(&colsum[b * 8 + tid], scs[tid]);
}

// ---------------------------------------------------------------------------
// K3c: updates[b,s,d] = sum_n (attn_vis+eps)/colsum * vv ; one block per b
// wave = one slot s, lane owns 4 consecutive d (64-bit f16 loads)
// ---------------------------------------------------------------------------
__global__ __launch_bounds__(256)
void k_updates(const float* __restrict__ attn_out, const _Float16* __restrict__ vvh,
               const float* __restrict__ colsum, float* __restrict__ updates)
{
    __shared__ float la[256 * 8];
    const int tid = threadIdx.x;
    const int b = blockIdx.x;
    const int s = tid >> 5;              // slot handled by this wave
    const int d4 = (tid & 31) * 4;       // 4 consecutive channels
    float acc[4] = {};

    for (int nc = 0; nc < 16; ++nc) {
        for (int idx = tid; idx < 2048; idx += 256)
            la[idx] = attn_out[((long)b * NN + nc * 256 + (idx >> 3)) * 8 + (idx & 7)] + EPSA;
        __syncthreads();
        if (nc < 15)
            __builtin_prefetch(vvh + ((long)b * NN + (nc + 1) * 256) * 128 + d4, 0, 0);
        for (int r = 0; r < 256; ++r) {
            const v4h v4 = *(const v4h*)(vvh + ((long)b * NN + nc * 256 + r) * 128 + d4);
            const float a = la[r * 8 + s];
            #pragma unroll
            for (int j = 0; j < 4; ++j) acc[j] += a * (float)v4[j];
        }
        __syncthreads();
    }
    const float inv = 1.f / colsum[b * 8 + s];
    #pragma unroll
    for (int j = 0; j < 4; ++j)
        updates[(b * 8 + s) * 128 + d4 + j] = acc[j] * inv;
}

// ---------------------------------------------------------------------------
// K3d: GRU_e + (MLP_e), GRU_i + (MLP_i), logits = upd @ p^T / 8
// one block per b; wave g handles row s=g; lanes split 64 elems (2 each)
// ---------------------------------------------------------------------------
__global__ __launch_bounds__(256)
void k_gru_mlp(const float* __restrict__ updates,
               float* __restrict__ slots_ext, const float* __restrict__ slots_int,
               float* __restrict__ logits_pi,
               const float* __restrict__ giWih, const float* __restrict__ giWhh,
               const float* __restrict__ gibih, const float* __restrict__ gibhh,
               const float* __restrict__ geWih, const float* __restrict__ geWhh,
               const float* __restrict__ gebih, const float* __restrict__ gebhh,
               const float* __restrict__ lnmig, const float* __restrict__ lnmib,
               const float* __restrict__ miW1, const float* __restrict__ mib1,
               const float* __restrict__ miW2, const float* __restrict__ mib2,
               const float* __restrict__ lnmeg, const float* __restrict__ lnmeb,
               const float* __restrict__ meW1, const float* __restrict__ meb1,
               const float* __restrict__ meW2, const float* __restrict__ meb2,
               const float* __restrict__ p,
               int do_mlp, int final_iter,
               float* __restrict__ out_slots, float* __restrict__ out_logits)
{
    __shared__ float ru[8 * 128];
    __shared__ float hext[8 * 64];
    __shared__ float hint[8 * 64];
    __shared__ float nxt[8 * 64];
    __shared__ float hln[8 * 64];
    __shared__ float m1[8 * 256];

    const int tid = threadIdx.x, b = blockIdx.x;
    const int s = tid >> 5, lane = tid & 31;

    for (int idx = tid; idx < 8 * 128; idx += 256)
        ru[idx] = updates[b * 1024 + idx];
    for (int idx = tid; idx < 8 * 64; idx += 256) {
        hext[idx] = slots_ext[b * 512 + idx];
        hint[idx] = slots_int[b * 512 + idx];
    }
    __syncthreads();

    // ================= EXT path =================
    {
        const float* xr = &ru[s * 128 + 64];
        const float* hr = &hext[s * 64];
        #pragma unroll
        for (int half = 0; half < 2; ++half) {
            const int e = lane + half * 32;
            float gir = gebih[e], giz = gebih[64 + e], gin = gebih[128 + e];
            float ghr = gebhh[e], ghz = gebhh[64 + e], ghn = gebhh[128 + e];
            for (int k = 0; k < 64; ++k) {
                const float xv = xr[k], hv = hr[k];
                gir += xv * geWih[k * 192 + e];
                giz += xv * geWih[k * 192 + 64 + e];
                gin += xv * geWih[k * 192 + 128 + e];
                ghr += hv * geWhh[k * 192 + e];
                ghz += hv * geWhh[k * 192 + 64 + e];
                ghn += hv * geWhh[k * 192 + 128 + e];
            }
            const float r = 1.f / (1.f + expf(-(gir + ghr)));
            const float z = 1.f / (1.f + expf(-(giz + ghz)));
            const float nn = tanhf(gin + r * ghn);
            nxt[s * 64 + e] = (1.f - z) * nn + z * hr[e];
        }
    }
    __syncthreads();
    if (do_mlp) {
        float a0 = nxt[s * 64 + lane], a1 = nxt[s * 64 + lane + 32];
        float ssum = a0 + a1, ssq = a0 * a0 + a1 * a1;
        #pragma unroll
        for (int off = 16; off > 0; off >>= 1) {
            ssum += __shfl_xor(ssum, off, 32);
            ssq  += __shfl_xor(ssq,  off, 32);
        }
        const float mean = ssum / 64.f;
        const float inv = rsqrtf(ssq / 64.f - mean * mean + 1e-5f);
        hln[s * 64 + lane]      = (a0 - mean) * inv * lnmeg[lane] + lnmeb[lane];
        hln[s * 64 + lane + 32] = (a1 - mean) * inv * lnmeg[lane + 32] + lnmeb[lane + 32];
        __syncthreads();
        for (int jj = 0; jj < 8; ++jj) {
            const int j = lane * 8 + jj;
            float acc = meb1[j];
            for (int k = 0; k < 64; ++k) acc += hln[s * 64 + k] * meW1[k * 256 + j];
            m1[s * 256 + j] = fmaxf(acc, 0.f);
        }
        __syncthreads();
        #pragma unroll
        for (int half = 0; half < 2; ++half) {
            const int e = lane + half * 32;
            float acc = meb2[e];
            for (int j = 0; j < 256; ++j) acc += m1[s * 256 + j] * meW2[j * 64 + e];
            nxt[s * 64 + e] += acc;
        }
        __syncthreads();
    }
    #pragma unroll
    for (int half = 0; half < 2; ++half) {
        const int e = lane + half * 32;
        const float v = nxt[s * 64 + e];
        slots_ext[b * 512 + s * 64 + e] = v;
        if (final_iter && s >= 1) out_slots[(b * 7 + (s - 1)) * 64 + e] = v;
    }
    __syncthreads();

    // ================= INT path =================
    {
        const float* xr = &ru[s * 128];
        const float* hr = &hint[s * 64];
        #pragma unroll
        for (int half = 0; half < 2; ++half) {
            const int e = lane + half * 32;
            float gir = gibih[e], giz = gibih[64 + e], gin = gibih[128 + e];
            float ghr = gibhh[e], ghz = gibhh[64 + e], ghn = gibhh[128 + e];
            for (int k = 0; k < 64; ++k) {
                const float xv = xr[k], hv = hr[k];
                gir += xv * giWih[k * 192 + e];
                giz += xv * giWih[k * 192 + 64 + e];
                gin += xv * giWih[k * 192 + 128 + e];
                ghr += hv * giWhh[k * 192 + e];
                ghz += hv * giWhh[k * 192 + 64 + e];
                ghn += hv * giWhh[k * 192 + 128 + e];
            }
            const float r = 1.f / (1.f + expf(-(gir + ghr)));
            const float z = 1.f / (1.f + expf(-(giz + ghz)));
            const float nn = tanhf(gin + r * ghn);
            nxt[s * 64 + e] = (1.f - z) * nn + z * hr[e];
        }
    }
    __syncthreads();
    if (do_mlp) {
        float a0 = nxt[s * 64 + lane], a1 = nxt[s * 64 + lane + 32];
        float ssum = a0 + a1, ssq = a0 * a0 + a1 * a1;
        #pragma unroll
        for (int off = 16; off > 0; off >>= 1) {
            ssum += __shfl_xor(ssum, off, 32);
            ssq  += __shfl_xor(ssq,  off, 32);
        }
        const float mean = ssum / 64.f;
        const float inv = rsqrtf(ssq / 64.f - mean * mean + 1e-5f);
        hln[s * 64 + lane]      = (a0 - mean) * inv * lnmig[lane] + lnmib[lane];
        hln[s * 64 + lane + 32] = (a1 - mean) * inv * lnmig[lane + 32] + lnmib[lane + 32];
        __syncthreads();
        for (int jj = 0; jj < 8; ++jj) {
            const int j = lane * 8 + jj;
            float acc = mib1[j];
            for (int k = 0; k < 64; ++k) acc += hln[s * 64 + k] * miW1[k * 256 + j];
            m1[s * 256 + j] = fmaxf(acc, 0.f);
        }
        __syncthreads();
        #pragma unroll
        for (int half = 0; half < 2; ++half) {
            const int e = lane + half * 32;
            float acc = mib2[e];
            for (int j = 0; j < 256; ++j) acc += m1[s * 256 + j] * miW2[j * 64 + e];
            nxt[s * 64 + e] += acc;
        }
        __syncthreads();
    }
    __syncthreads();
    // logits_pi = int_upd @ p^T * 1/8
    #pragma unroll
    for (int half = 0; half < 2; ++half) {
        const int c = lane + half * 32;
        float acc = 0.f;
        for (int e = 0; e < 64; ++e) acc += nxt[s * 64 + e] * p[c * 64 + e];
        acc *= 0.125f;
        logits_pi[b * 512 + s * 64 + c] = acc;
        if (final_iter && s >= 1) out_logits[(b * 7 + (s - 1)) * 64 + c] = acc;
    }
}

// ---------------------------------------------------------------------------
extern "C" void kernel_launch(void* const* d_in, const int* in_sizes, int n_in,
                              void* d_out, int out_size, void* d_ws, size_t ws_size,
                              hipStream_t stream)
{
    // setup_inputs() dict order
    const float* slots_bck = (const float*)d_in[0];
    const float* inputs    = (const float*)d_in[1];
    const float* protos    = (const float*)d_in[2];
    const float* ext_noise = (const float*)d_in[3];
    const float* int_noise = (const float*)d_in[4];
    const float* gumbel    = (const float*)d_in[5];
    const int*   tau       = (const int*)  d_in[6];
    const float* ext_mu    = (const float*)d_in[7];
    const float* ext_ls    = (const float*)d_in[8];
    const float* int_mu    = (const float*)d_in[9];
    const float* int_ls    = (const float*)d_in[10];
    const float* ln_in_g   = (const float*)d_in[11];
    const float* ln_in_b   = (const float*)d_in[12];
    const float* ln_p_g    = (const float*)d_in[13];
    const float* ln_p_b    = (const float*)d_in[14];
    const float* ln_mi_g   = (const float*)d_in[15];
    const float* ln_mi_b   = (const float*)d_in[16];
    const float* ln_me_g   = (const float*)d_in[17];
    const float* ln_me_b   = (const float*)d_in[18];
    const float* Wq        = (const float*)d_in[19];
    const float* Wk        = (const float*)d_in[20];
    const float* Wv        = (const float*)d_in[21];
    const float* Wp        = (const float*)d_in[22];
    const float* gi_Wih    = (const float*)d_in[23];
    const float* gi_Whh    = (const float*)d_in[24];
    const float* gi_bih    = (const float*)d_in[25];
    const float* gi_bhh    = (const float*)d_in[26];
    const float* ge_Wih    = (const float*)d_in[27];
    const float* ge_Whh    = (const float*)d_in[28];
    const float* ge_bih    = (const float*)d_in[29];
    const float* ge_bhh    = (const float*)d_in[30];
    const float* mi_W1     = (const float*)d_in[31];
    const float* mi_b1     = (const float*)d_in[32];
    const float* mi_W2     = (const float*)d_in[33];
    const float* mi_b2     = (const float*)d_in[34];
    const float* me_W1     = (const float*)d_in[35];
    const float* me_b1     = (const float*)d_in[36];
    const float* me_W2     = (const float*)d_in[37];
    const float* me_b2     = (const float*)d_in[38];

    // d_out layout: slots_ext[:,1:] | logits_pi[:,1:] | attn_vis
    float* out        = (float*)d_out;
    float* out_slots  = out;
    float* out_logits = out + (size_t)BB * 7 * EEC;
    float* attn_out   = out + (size_t)2 * BB * 7 * EEC;

    // workspace layout
    char* ws = (char*)d_ws;
    const size_t MROWS = (size_t)BB * NN;            // 262144
    _Float16* kkh  = (_Float16*)ws;                         ws += MROWS * 128 * 2;
    _Float16* vvh  = (_Float16*)ws;                         ws += MROWS * 128 * 2;
    float* colsum       = (float*)ws;                       ws += 512 * 4;
    float* slots_ext_ws = (float*)ws;                       ws += 512 * 64 * 4;
    float* logits_pi_ws = (float*)ws;                       ws += 512 * 64 * 4;
    float* slots_int_ws = (float*)ws;                       ws += 512 * 64 * 4;
    float* q_ws         = (float*)ws;                       ws += 512 * 128 * 4;
    float* updates_ws   = (float*)ws;                       ws += 512 * 128 * 4;
    float* p_ws         = (float*)ws;                       ws += 64 * 64 * 4;

    // K1: LN + dual WMMA GEMM -> kkh, vvh
    k_ln_gemm<<<(int)(MROWS / 64), 128, 0, stream>>>(inputs, ln_in_g, ln_in_b,
                                                     Wk, Wv, kkh, vvh);
    // K2: proto projection
    k_protos<<<1, 64, 0, stream>>>(protos, ln_p_g, ln_p_b, Wp, p_ws);
    // init slots_ext / logits_pi
    k_init<<<128, 256, 0, stream>>>(ext_mu, ext_ls, ext_noise,
                                    int_mu, int_ls, int_noise,
                                    slots_ext_ws, logits_pi_ws);

    for (int i = 0; i < 3; ++i) {
        const float* gum_i = gumbel + (size_t)i * 512 * 64;
        k_slots_q<<<BB, 256, 0, stream>>>(logits_pi_ws, gum_i, tau, p_ws,
                                          slots_ext_ws, slots_bck, Wq,
                                          slots_int_ws, q_ws, colsum);
        k_attn<<<dim3(16, BB), 256, 0, stream>>>(kkh, q_ws, attn_out, colsum);
        k_updates<<<BB, 256, 0, stream>>>(attn_out, vvh, colsum, updates_ws);
        k_gru_mlp<<<BB, 256, 0, stream>>>(updates_ws,
                                          slots_ext_ws, slots_int_ws, logits_pi_ws,
                                          gi_Wih, gi_Whh, gi_bih, gi_bhh,
                                          ge_Wih, ge_Whh, ge_bih, ge_bhh,
                                          ln_mi_g, ln_mi_b, mi_W1, mi_b1, mi_W2, mi_b2,
                                          ln_me_g, ln_me_b, me_W1, me_b1, me_W2, me_b2,
                                          p_ws,
                                          (i < 2) ? 1 : 0, (i == 2) ? 1 : 0,
                                          out_slots, out_logits);
    }
}
